// BaseLayer_32160715112901
// MI455X (gfx1250) — matrix operand order repaired and checked by reading, not compiled, not run on hardware
//
#include <hip/hip_runtime.h>

#define E_CNT 8
#define M_DIM 512
#define H_DIM 1024
#define BM    64
#define SX    (M_DIM + 8)   // LDS row stride (bf16 elems) for X tile
#define SH    (H_DIM + 8)   // LDS row stride (bf16 elems) for H tile

typedef __bf16 v16bf __attribute__((ext_vector_type(16)));
typedef __bf16 v8bf  __attribute__((ext_vector_type(8)));
typedef float  v8f   __attribute__((ext_vector_type(8)));
typedef __attribute__((address_space(3))) char lds_char;

__device__ __forceinline__ __bf16 f2bf(float f) {
  unsigned u = __builtin_bit_cast(unsigned, f);
  u += 0x7fffu + ((u >> 16) & 1u);           // round-to-nearest-even
  unsigned short h = (unsigned short)(u >> 16);
  return __builtin_bit_cast(__bf16, h);
}

// ---------------------------------------------------------------------------
// Prep A: activations f32 -> bf16 (vectorized, 8 elems/thread/iter).
// ---------------------------------------------------------------------------
__global__ void moe_acvt_kernel(const float* __restrict__ src,
                                __bf16* __restrict__ dst, size_t n8) {
  size_t i = (size_t)blockIdx.x * blockDim.x + threadIdx.x;
  size_t stride = (size_t)gridDim.x * blockDim.x;
  for (; i < n8; i += stride) {
    const float4* s = (const float4*)(src + i * 8);
    float4 a = s[0], b = s[1];
    v8bf o;
    o[0] = f2bf(a.x); o[1] = f2bf(a.y); o[2] = f2bf(a.z); o[3] = f2bf(a.w);
    o[4] = f2bf(b.x); o[5] = f2bf(b.y); o[6] = f2bf(b.z); o[7] = f2bf(b.w);
    ((v8bf*)dst)[i] = o;
  }
}

// ---------------------------------------------------------------------------
// Prep B: weights f32 [e][K][N] -> bf16 transposed [e][N][K] (32x32 LDS tile,
// both global accesses coalesced).
// ---------------------------------------------------------------------------
__global__ __launch_bounds__(256) void moe_wcvt_kernel(
    const float* __restrict__ src, __bf16* __restrict__ dst, int K, int N) {
  __shared__ float tile[32][33];
  const float* s = src + (size_t)blockIdx.z * K * N;
  __bf16* d = dst + (size_t)blockIdx.z * K * N;
  const int n0 = blockIdx.x * 32, k0 = blockIdx.y * 32;
  const int tx = threadIdx.x & 31, ty = threadIdx.x >> 5;  // 32 x 8
#pragma unroll
  for (int i = 0; i < 4; ++i)
    tile[ty + 8 * i][tx] = s[(size_t)(k0 + ty + 8 * i) * N + n0 + tx];
  __syncthreads();
#pragma unroll
  for (int i = 0; i < 4; ++i)
    d[(size_t)(n0 + ty + 8 * i) * K + k0 + tx] = f2bf(tile[tx][ty + 8 * i]);
}

// ---------------------------------------------------------------------------
// Kernel 1: top-1 gate. One wave per token; wg staged in LDS. Deterministic
// per-block partial sums of softmax probs for l_aux.
// ---------------------------------------------------------------------------
__global__ __launch_bounds__(256) void moe_gate_kernel(
    const float* __restrict__ feats, const float* __restrict__ wg,
    int* __restrict__ idx, float* __restrict__ gate_max,
    float* __restrict__ me_part) {
  __shared__ float wgs[M_DIM * E_CNT];      // 16 KB
  __shared__ float gsum[8][E_CNT];
  const int tid  = threadIdx.x;
  const int wave = tid >> 5;
  const int lane = tid & 31;

  for (int i = tid; i < M_DIM * E_CNT; i += 256) wgs[i] = wg[i];
  __syncthreads();

  const int t = blockIdx.x * 8 + wave;
  const float* xr = feats + (size_t)t * M_DIM;
  float acc[E_CNT];
#pragma unroll
  for (int e = 0; e < E_CNT; ++e) acc[e] = 0.0f;

  for (int k = lane; k < M_DIM; k += 32) {
    float x = xr[k];
#pragma unroll
    for (int e = 0; e < E_CNT; ++e) acc[e] += x * wgs[k * E_CNT + e];
  }
#pragma unroll
  for (int e = 0; e < E_CNT; ++e) {
    float v = acc[e];
#pragma unroll
    for (int off = 16; off > 0; off >>= 1) v += __shfl_xor(v, off, 32);
    acc[e] = v;
  }

  if (lane == 0) {
    float mx = acc[0];
#pragma unroll
    for (int e = 1; e < E_CNT; ++e) mx = fmaxf(mx, acc[e]);
    float g[E_CNT], s = 0.0f;
#pragma unroll
    for (int e = 0; e < E_CNT; ++e) { g[e] = __expf(acc[e] - mx); s += g[e]; }
    float inv = 1.0f / s;
    float bg = -1.0f; int am = 0;
#pragma unroll
    for (int e = 0; e < E_CNT; ++e) {
      g[e] *= inv;
      if (g[e] > bg) { bg = g[e]; am = e; }
      gsum[wave][e] = g[e];
    }
    idx[t] = am;
    gate_max[t] = bg;
  }
  __syncthreads();
  if (tid < E_CNT) {
    float s = 0.0f;
#pragma unroll
    for (int w = 0; w < 8; ++w) s += gsum[w][tid];
    me_part[blockIdx.x * E_CNT + tid] = s;
  }
}

// ---------------------------------------------------------------------------
// Kernel 2: arrival-order capacity scan (single block, ballot + LDS scan).
// ---------------------------------------------------------------------------
__global__ __launch_bounds__(256) void moe_scan_kernel(
    const int* __restrict__ idx, const float* __restrict__ gate_max,
    int* __restrict__ slot_token, float* __restrict__ slot_gate,
    int* __restrict__ counts, int S, int cap) {
  __shared__ int wtot[8][E_CNT];
  __shared__ int wpre[8][E_CNT];
  __shared__ int base[E_CNT];
  const int tid  = threadIdx.x;
  const int wave = tid >> 5;
  const int lane = tid & 31;
  if (tid < E_CNT) base[tid] = 0;
  __syncthreads();

  for (int chunk = 0; chunk < S; chunk += 256) {
    const int t = chunk + tid;
    const int e = idx[t];
    const unsigned lt = (1u << lane) - 1u;
    int within = 0;
#pragma unroll
    for (int ee = 0; ee < E_CNT; ++ee) {
      unsigned bm = (unsigned)__ballot(e == ee);
      if (e == ee) within = __popc(bm & lt);
      if (lane == 0) wtot[wave][ee] = __popc(bm);
    }
    __syncthreads();
    if (tid < E_CNT) {
      int off = base[tid];
#pragma unroll
      for (int w = 0; w < 8; ++w) { wpre[w][tid] = off; off += wtot[w][tid]; }
      base[tid] = off;
    }
    __syncthreads();
    const int loc = wpre[wave][e] + within;
    if (loc < cap) {
      slot_token[e * cap + loc] = t;
      slot_gate[e * cap + loc]  = gate_max[t];
    }
    __syncthreads();
  }
  if (tid < E_CNT) counts[tid] = base[tid] < cap ? base[tid] : cap;
}

// ---------------------------------------------------------------------------
// Kernel 3: zero output.
// ---------------------------------------------------------------------------
__global__ void moe_zero_kernel(float* __restrict__ p, size_t n) {
  size_t i = (size_t)blockIdx.x * blockDim.x + threadIdx.x;
  size_t stride = (size_t)gridDim.x * blockDim.x;
  for (; i < n; i += stride) p[i] = 0.0f;
}

// ---------------------------------------------------------------------------
// Kernel 4: l_aux (deterministic fixed-order reduction).
// ---------------------------------------------------------------------------
__global__ void moe_laux_kernel(const float* __restrict__ me_part,
                                const int* __restrict__ counts,
                                float* __restrict__ out_laux, int nblocks,
                                int S) {
  __shared__ float me[E_CNT];
  const int tid = threadIdx.x;
  if (tid < E_CNT) {
    float s = 0.0f;
    for (int b = 0; b < nblocks; ++b) s += me_part[b * E_CNT + tid];
    me[tid] = s;
  }
  __syncthreads();
  if (tid == 0) {
    const float invS = 1.0f / (float)S;
    float l = 0.0f;
#pragma unroll
    for (int e = 0; e < E_CNT; ++e)
      l += (me[e] * invS) * ((float)counts[e] * invS);
    out_laux[0] = l * (float)E_CNT;
  }
}

// ---------------------------------------------------------------------------
// Kernel 5: per-expert FFN via v_wmma_f32_16x16x32_bf16.
//  - X tile staged with async global->LDS (b128) from pre-converted bf16 acts
//  - B fragments: two global_load_b128 from transposed bf16 weights [e][N][K]
//  - A fragments: two ds_load_b128 (frag halves are contiguous K runs)
// Block = 256 threads (8 waves), BM=64 token rows of one expert.
// ---------------------------------------------------------------------------
__global__ __launch_bounds__(256, 1) void moe_ffn_kernel(
    const __bf16* __restrict__ featsbf, const __bf16* __restrict__ w1t,
    const float* __restrict__ b1, const __bf16* __restrict__ w2t,
    const float* __restrict__ b2, const int* __restrict__ slot_token,
    const float* __restrict__ slot_gate, const int* __restrict__ counts,
    float* __restrict__ out, int cap) {
  extern __shared__ char smem[];
  __bf16* Xs = (__bf16*)smem;                                       // BM x SX
  __bf16* Hs = (__bf16*)(smem + (size_t)BM * SX * sizeof(__bf16));  // BM x SH
  int*   tok = (int*)(smem + (size_t)BM * (SX + SH) * sizeof(__bf16));
  float* gts = (float*)(tok + BM);

  const int tiles = cap / BM;
  const int e     = blockIdx.x / tiles;
  const int row0  = (blockIdx.x % tiles) * BM;
  const int cnt   = counts[e];
  if (row0 >= cnt) return;   // uniform: whole tile empty

  const int tid  = threadIdx.x;
  const int wave = tid >> 5;
  const int lane = tid & 31;
  const int col  = lane & 15;
  const int hi   = lane >> 4;

  if (tid < BM) {
    const bool valid = (row0 + tid) < cnt;
    tok[tid] = valid ? slot_token[e * cap + row0 + tid] : -1;
    gts[tid] = valid ? slot_gate[e * cap + row0 + tid] : 0.0f;
  }
  __syncthreads();

  // ---- async-stage X tile (bf16 rows, 16B chunks) into LDS ----
  {
    const unsigned xs_base = (unsigned)(size_t)(lds_char*)Xs;
    const int chunks = BM * (M_DIM / 8);   // 4096 x 16B
    for (int c = tid; c < chunks; c += 256) {
      const int r  = c >> 6;               // 64 chunks per row
      const int cc = (c & 63) * 8;
      const int t  = tok[r];
      if (t >= 0) {
        const __bf16* g = featsbf + (size_t)t * M_DIM + cc;
        const unsigned loff = xs_base + (unsigned)((r * SX + cc) * 2);
        asm volatile("global_load_async_to_lds_b128 %0, %1, off"
                     :: "v"(loff), "v"((unsigned long long)(size_t)g)
                     : "memory");
      }
    }
    asm volatile("s_wait_asynccnt 0x0" ::: "memory");
  }
  __syncthreads();

  // ---- GEMM1: H = relu(X @ W1 + b1) ; wave owns H cols [wave*128, +128) ----
  const __bf16* w1te = w1t + (size_t)e * M_DIM * H_DIM;  // [n][k], K = M_DIM
  for (int cb = 0; cb < 8; ++cb) {
    const int nn = wave * 128 + cb * 16 + col;
    v8f acc[4] = {{0}, {0}, {0}, {0}};
    for (int k0 = 0; k0 < M_DIM; k0 += 32) {
      const v8bf* bp = (const v8bf*)(w1te + (size_t)nn * M_DIM + k0 + hi * 16);
      const v16bf bfrag = __builtin_shufflevector(
          bp[0], bp[1], 0, 1, 2, 3, 4, 5, 6, 7, 8, 9, 10, 11, 12, 13, 14, 15);
#pragma unroll
      for (int rb = 0; rb < 4; ++rb) {
        const v8bf* xp = (const v8bf*)(Xs + (rb * 16 + col) * SX + k0 + hi * 16);
        const v16bf afrag = __builtin_shufflevector(
            xp[0], xp[1], 0, 1, 2, 3, 4, 5, 6, 7, 8, 9, 10, 11, 12, 13, 14, 15);
        acc[rb] = __builtin_amdgcn_wmma_f32_16x16x32_bf16(
            false, afrag, false, bfrag, (short)0, acc[rb], false, false);
      }
    }
    const float bias = b1[e * H_DIM + nn];
#pragma unroll
    for (int rb = 0; rb < 4; ++rb) {
#pragma unroll
      for (int j = 0; j < 8; ++j) {       // C/D: VGPR j -> row j + 8*hi
        const int row = rb * 16 + hi * 8 + j;
        const float hv = acc[rb][j] + bias;
        Hs[row * SH + nn] = f2bf(hv > 0.0f ? hv : 0.0f);
      }
    }
  }
  __syncthreads();

  // ---- GEMM2: Y = gate * (H @ W2 + b2) ; wave owns Y cols [wave*64, +64) ----
  const __bf16* w2te = w2t + (size_t)e * H_DIM * M_DIM;  // [n][k], K = H_DIM
  for (int cb = 0; cb < 4; ++cb) {
    const int nn = wave * 64 + cb * 16 + col;
    v8f acc[4] = {{0}, {0}, {0}, {0}};
    for (int k0 = 0; k0 < H_DIM; k0 += 32) {
      const v8bf* bp = (const v8bf*)(w2te + (size_t)nn * H_DIM + k0 + hi * 16);
      const v16bf bfrag = __builtin_shufflevector(
          bp[0], bp[1], 0, 1, 2, 3, 4, 5, 6, 7, 8, 9, 10, 11, 12, 13, 14, 15);
#pragma unroll
      for (int rb = 0; rb < 4; ++rb) {
        const v8bf* hp = (const v8bf*)(Hs + (rb * 16 + col) * SH + k0 + hi * 16);
        const v16bf afrag = __builtin_shufflevector(
            hp[0], hp[1], 0, 1, 2, 3, 4, 5, 6, 7, 8, 9, 10, 11, 12, 13, 14, 15);
        acc[rb] = __builtin_amdgcn_wmma_f32_16x16x32_bf16(
            false, afrag, false, bfrag, (short)0, acc[rb], false, false);
      }
    }
    const float bias2 = b2[e * M_DIM + nn];
#pragma unroll
    for (int rb = 0; rb < 4; ++rb) {
#pragma unroll
      for (int j = 0; j < 8; ++j) {
        const int row = rb * 16 + hi * 8 + j;
        const int t = tok[row];
        if (t >= 0)
          out[(size_t)t * M_DIM + nn] = gts[row] * (acc[rb][j] + bias2);
      }
    }
  }
}

// ---------------------------------------------------------------------------
extern "C" void kernel_launch(void* const* d_in, const int* in_sizes, int n_in,
                              void* d_out, int out_size, void* d_ws,
                              size_t ws_size, hipStream_t stream) {
  const float* feats = (const float*)d_in[0];  // [S, M]
  const float* wg    = (const float*)d_in[1];  // [M, E]
  const float* w1    = (const float*)d_in[2];  // [E, M, H]
  const float* b1    = (const float*)d_in[3];  // [E, H]
  const float* w2    = (const float*)d_in[4];  // [E, H, M]
  const float* b2    = (const float*)d_in[5];  // [E, M]
  float* out = (float*)d_out;

  const int S   = in_sizes[0] / M_DIM;  // 8192
  const int cap = S / E_CNT;            // 1024

  char* ws = (char*)d_ws;
  int*   idx        = (int*)ws;   ws += (size_t)S * sizeof(int);
  float* gate_max   = (float*)ws; ws += (size_t)S * sizeof(float);
  float* me_part    = (float*)ws; ws += (size_t)(S / 8) * E_CNT * sizeof(float);
  int*   slot_token = (int*)ws;   ws += (size_t)E_CNT * cap * sizeof(int);
  float* slot_gate  = (float*)ws; ws += (size_t)E_CNT * cap * sizeof(float);
  int*   counts     = (int*)ws;   ws += 256;
  __bf16* featsbf   = (__bf16*)ws; ws += (size_t)S * M_DIM * sizeof(__bf16);
  __bf16* w1t       = (__bf16*)ws; ws += (size_t)E_CNT * M_DIM * H_DIM * sizeof(__bf16);
  __bf16* w2t       = (__bf16*)ws; ws += (size_t)E_CNT * M_DIM * H_DIM * sizeof(__bf16);

  // activation & weight conversion (bf16; weights transposed to [e][N][K])
  moe_acvt_kernel<<<dim3(1024), dim3(256), 0, stream>>>(
      feats, featsbf, (size_t)S * M_DIM / 8);
  moe_wcvt_kernel<<<dim3(H_DIM / 32, M_DIM / 32, E_CNT), dim3(256), 0, stream>>>(
      w1, w1t, M_DIM, H_DIM);
  moe_wcvt_kernel<<<dim3(M_DIM / 32, H_DIM / 32, E_CNT), dim3(256), 0, stream>>>(
      w2, w2t, H_DIM, M_DIM);

  moe_gate_kernel<<<dim3(S / 8), dim3(256), 0, stream>>>(feats, wg, idx,
                                                         gate_max, me_part);
  moe_scan_kernel<<<dim3(1), dim3(256), 0, stream>>>(idx, gate_max, slot_token,
                                                     slot_gate, counts, S, cap);
  moe_zero_kernel<<<dim3(1024), dim3(256), 0, stream>>>(out, (size_t)out_size);
  moe_laux_kernel<<<dim3(1), dim3(64), 0, stream>>>(
      me_part, counts, out + (out_size - 1), S / 8, S);

  const size_t smem =
      (size_t)BM * (SX + SH) * sizeof(__bf16) + BM * (sizeof(int) + sizeof(float));
  moe_ffn_kernel<<<dim3(E_CNT * (cap / BM)), dim3(256), smem, stream>>>(
      featsbf, w1t, b1, w2t, b2, slot_token, slot_gate, counts, out, cap);
}
